// MultiHeadLatentAttention_50500225466824
// MI455X (gfx1250) — compile-verified, matrix operand order checked by reading, not run on hardware
//
#include <hip/hip_runtime.h>

#define B   4
#define S   1024
#define HID 2048
#define NH  16
#define LAT 512
#define HD  128

typedef __bf16 v16bf __attribute__((ext_vector_type(16)));
typedef __bf16 v4bf  __attribute__((ext_vector_type(4)));
typedef float  v8f   __attribute__((ext_vector_type(8)));

static __device__ inline v8f wmma_bf16(v16bf a, v16bf b, v8f c) {
  // D = A(16x32 bf16) x B(32x16 bf16) + C(16x16 f32)
  return __builtin_amdgcn_wmma_f32_16x16x32_bf16(false, a, false, b, (short)0, c,
                                                 false, false);
}

// Async DMA: copy 16B from global to LDS, no VGPR round-trip (ASYNCcnt).
static __device__ inline void async_g2l_b128(__bf16* lds, const __bf16* g) {
  asm volatile("global_load_async_to_lds_b128 %0, %1, off"
               :
               : "v"((unsigned)(unsigned long long)lds), "v"(g)
               : "memory");
}
static __device__ inline void wait_async0() {
#if __has_builtin(__builtin_amdgcn_s_wait_asynccnt)
  __builtin_amdgcn_s_wait_asynccnt(0);
#else
  asm volatile("s_wait_asynccnt 0x0" ::: "memory");
#endif
}

// 16x32 A-operand fragment (also used for B by staging B^T). ISA 7.12.2:
// lanes 0-15: M=row, K base 0; lanes 16-31: same rows, K base 8;
// elems 0..7 -> K+0..7, elems 8..15 -> K+16..23.  With a 16B-aligned row
// stride this compiles to 2x ds_load_b128 per fragment.
static __device__ inline v16bf load_frag(const __bf16* src, int ld) {
  const int lane = threadIdx.x & 31;
  const int row  = lane & 15;
  const int kb   = (lane >> 4) << 3;
  v16bf f;
#pragma unroll
  for (int j = 0; j < 16; ++j) {
    const int k = kb + (j < 8 ? j : j + 8);
    f[j] = src[row * ld + k];
  }
  return f;
}

static __device__ inline float red_max16(float v) {
#pragma unroll
  for (int m = 8; m >= 1; m >>= 1) v = fmaxf(v, __shfl_xor(v, m, 32));
  return v;
}
static __device__ inline float red_sum16(float v) {
#pragma unroll
  for (int m = 8; m >= 1; m >>= 1) v += __shfl_xor(v, m, 32);
  return v;
}

// ---------------------------------------------------------------------------
// One-shot f32 -> bf16 conversion (vectorized, 4 elems/thread).
// ---------------------------------------------------------------------------
__global__ __launch_bounds__(256) void cvt_bf16_kernel(
    const float* __restrict__ in, __bf16* __restrict__ out, int n4) {
  const int idx = blockIdx.x * 256 + threadIdx.x;
  if (idx < n4) {
    const float4 v = ((const float4*)in)[idx];
    v4bf o;
    o[0] = (__bf16)v.x; o[1] = (__bf16)v.y; o[2] = (__bf16)v.z; o[3] = (__bf16)v.w;
    ((v4bf*)out)[idx] = o;
  }
}

// ---------------------------------------------------------------------------
// f32 (K x N) -> bf16 transposed (N x K), 64x64 LDS tile per block.
// ---------------------------------------------------------------------------
__global__ __launch_bounds__(256) void cvt_t_bf16_kernel(
    const float* __restrict__ in, __bf16* __restrict__ out, int K, int N) {
  __shared__ float tile[64][65];
  const int k0 = blockIdx.y * 64, n0 = blockIdx.x * 64;
#pragma unroll
  for (int i = 0; i < 16; ++i) {
    const int idx = threadIdx.x + i * 256;
    const int r = idx >> 6, c = idx & 63;
    tile[r][c] = in[(size_t)(k0 + r) * N + n0 + c];
  }
  __syncthreads();
#pragma unroll
  for (int i = 0; i < 4; ++i) {
    const int idx = threadIdx.x + i * 256;
    const int n = idx >> 4, k4 = (idx & 15) * 4;
    v4bf o;
#pragma unroll
    for (int j = 0; j < 4; ++j) o[j] = (__bf16)tile[k4 + j][n];
    *(v4bf*)&out[(size_t)(n0 + n) * K + k0 + k4] = o;
  }
}

// ---------------------------------------------------------------------------
// GEMM: C[M,N] = A[M,K] x Bt[N,K]^T (+bias). bf16 operands, f32 accumulate.
// Block 256 thr (8 waves), tile 128x64, BK=64; wave computes 32x32 (8 WMMA /
// K-step).  Staging: async b128 DMA to LDS, double-buffered so the next
// tile's DMA overlaps the current tile's WMMAs.
// out_mode: 0 = bf16 row-major, 1 = bf16 scattered as V^T (b,h,d,s),
//           2 = f32 row-major.
// ---------------------------------------------------------------------------
__global__ __launch_bounds__(256) void gemm_bf16_kernel(
    const __bf16* __restrict__ A, const __bf16* __restrict__ Bt,
    const float* __restrict__ bias, __bf16* __restrict__ Cb,
    float* __restrict__ Cf, int M, int N, int K, int out_mode) {
  __shared__ __bf16 Asm[2][128 * 72];  // [row][k], 144B stride (16B multiple)
  __shared__ __bf16 Bsm[2][64 * 72];   // [n][k]

  const int m0   = blockIdx.y * 128;
  const int n0   = blockIdx.x * 64;
  const int w    = threadIdx.x >> 5;
  const int lane = threadIdx.x & 31;
  const int wm   = (w >> 1) * 32;
  const int wn   = (w & 1) * 32;

  auto stage = [&](int kk, int buf) {
#pragma unroll
    for (int i = 0; i < 4; ++i) {  // A tile: 128 rows x 64 k
      const int idx = threadIdx.x + i * 256;
      const int r = idx >> 3, c8 = (idx & 7) * 8;
      async_g2l_b128(&Asm[buf][r * 72 + c8],
                     &A[(size_t)(m0 + r) * K + kk + c8]);
    }
#pragma unroll
    for (int i = 0; i < 2; ++i) {  // B tile: 64 rows x 64 k
      const int idx = threadIdx.x + i * 256;
      const int r = idx >> 3, c8 = (idx & 7) * 8;
      async_g2l_b128(&Bsm[buf][r * 72 + c8],
                     &Bt[(size_t)(n0 + r) * K + kk + c8]);
    }
  };

  v8f acc[2][2] = {};

  stage(0, 0);
  wait_async0();
  __syncthreads();

  int cur = 0;
  for (int kk = 0; kk < K; kk += 64) {
    if (kk + 64 < K) stage(kk + 64, cur ^ 1);  // DMA next tile during compute

#pragma unroll
    for (int kc = 0; kc < 64; kc += 32) {
      const v16bf af0 = load_frag(&Asm[cur][(wm + 0) * 72 + kc], 72);
      const v16bf af1 = load_frag(&Asm[cur][(wm + 16) * 72 + kc], 72);
      const v16bf bf0 = load_frag(&Bsm[cur][(wn + 0) * 72 + kc], 72);
      const v16bf bf1 = load_frag(&Bsm[cur][(wn + 16) * 72 + kc], 72);
      acc[0][0] = wmma_bf16(af0, bf0, acc[0][0]);
      acc[0][1] = wmma_bf16(af0, bf1, acc[0][1]);
      acc[1][0] = wmma_bf16(af1, bf0, acc[1][0]);
      acc[1][1] = wmma_bf16(af1, bf1, acc[1][1]);
    }
    wait_async0();
    __syncthreads();
    cur ^= 1;
  }

  // Epilogue. C-layout: VGPR r -> M = r + 8*(lane>=16), N = lane&15.
#pragma unroll
  for (int i = 0; i < 2; ++i)
#pragma unroll
    for (int j = 0; j < 2; ++j)
#pragma unroll
      for (int r = 0; r < 8; ++r) {
        const int mm = m0 + wm + i * 16 + r + ((lane >> 4) << 3);
        const int nn = n0 + wn + j * 16 + (lane & 15);
        float o = acc[i][j][r];
        if (bias) o += bias[nn];
        if (out_mode == 2) {
          Cf[(size_t)mm * N + nn] = o;
        } else if (out_mode == 1) {  // V^T: (b, h, d, s) for attention B-op
          const int b = mm >> 10, s = mm & (S - 1);
          const int h = nn >> 7, d = nn & (HD - 1);
          Cb[(((size_t)b * NH + h) * HD + d) * S + s] = (__bf16)o;
        } else {
          Cb[(size_t)mm * N + nn] = (__bf16)o;
        }
      }
}

// ---------------------------------------------------------------------------
// RoPE in-place on bf16 q and k. Thread handles 4 consecutive i of one
// (b,s,h); pairs (i, i+64) rotated together.
// ---------------------------------------------------------------------------
union BF4 { uint2 u; __bf16 h[4]; };

__global__ __launch_bounds__(256) void rope_kernel(__bf16* __restrict__ q,
                                                   __bf16* __restrict__ k) {
  const int idx = blockIdx.x * 256 + threadIdx.x;  // over B*S*NH*16
  const int i4 = (idx & 15) * 4;
  const int t = idx >> 4;
  const int h = t & (NH - 1);
  const int bs = t >> 4;               // b*S + s
  const int s = bs & (S - 1);
  const size_t base = (size_t)bs * HID + h * HD + i4;

  float c[4], sn[4];
#pragma unroll
  for (int j = 0; j < 4; ++j) {
    const float inv_freq = __powf(10000.0f, -(float)(i4 + j) * (1.0f / 64.0f));
    const float ang = (float)s * inv_freq;
    c[j] = __cosf(ang); sn[j] = __sinf(ang);
  }
  {
    BF4 lo, hi;
    lo.u = *(const uint2*)&q[base];
    hi.u = *(const uint2*)&q[base + 64];
#pragma unroll
    for (int j = 0; j < 4; ++j) {
      const float x1 = (float)lo.h[j], x2 = (float)hi.h[j];
      lo.h[j] = (__bf16)(x1 * c[j] - x2 * sn[j]);
      hi.h[j] = (__bf16)(x2 * c[j] + x1 * sn[j]);
    }
    *(uint2*)&q[base] = lo.u;
    *(uint2*)&q[base + 64] = hi.u;
  }
  {
    BF4 lo, hi;
    lo.u = *(const uint2*)&k[base];
    hi.u = *(const uint2*)&k[base + 64];
#pragma unroll
    for (int j = 0; j < 4; ++j) {
      const float x1 = (float)lo.h[j], x2 = (float)hi.h[j];
      lo.h[j] = (__bf16)(x1 * c[j] - x2 * sn[j]);
      hi.h[j] = (__bf16)(x2 * c[j] + x1 * sn[j]);
    }
    *(uint2*)&k[base] = lo.u;
    *(uint2*)&k[base + 64] = hi.u;
  }
}

// ---------------------------------------------------------------------------
// Causal flash attention. Grid (B*NH, S/128), block 256 = 8 waves; wave owns
// 16 query rows. K tiles row-major, V pre-transposed (b,h,d,s); both staged
// via async b128 DMA, double-buffered against the WMMA/softmax compute.
// ---------------------------------------------------------------------------
__global__ __launch_bounds__(256) void attn_kernel(
    const __bf16* __restrict__ q, const __bf16* __restrict__ k,
    const __bf16* __restrict__ vt, __bf16* __restrict__ ctx) {
  __shared__ __bf16 Ksm[2][32 * 136];     // [key][d], 272B stride
  __shared__ __bf16 Vsm[2][128 * 40];     // [d][key], 80B stride
  __shared__ __bf16 Psm[8 * 16 * 40];     // per-wave P scratch [m][key]

  const int bh   = blockIdx.x;
  const int b    = bh / NH;
  const int h    = bh % NH;
  const int q0b  = blockIdx.y * 128;
  const int w    = threadIdx.x >> 5;
  const int lane = threadIdx.x & 31;
  const int q0w  = q0b + w * 16;
  __bf16* pw = &Psm[w * 16 * 40];
  const float scale = 0.08838834764831845f;  // 1/sqrt(128)

  const __bf16* qbase = q + ((size_t)b * S + q0w) * HID + (size_t)h * HD;
  v16bf qf[4];
#pragma unroll
  for (int kc = 0; kc < 4; ++kc) qf[kc] = load_frag(qbase + kc * 32, HID);

  const size_t vbase = ((size_t)b * NH + h) * (size_t)HD * S;

  auto stage = [&](int kb, int buf) {
#pragma unroll
    for (int i = 0; i < 2; ++i) {  // K tile (32 keys x 128 d)
      const int idx = threadIdx.x + i * 256;
      const int kk = idx >> 4, c8 = (idx & 15) * 8;
      async_g2l_b128(&Ksm[buf][kk * 136 + c8],
                     &k[((size_t)b * S + kb + kk) * HID + h * HD + c8]);
    }
#pragma unroll
    for (int i = 0; i < 2; ++i) {  // V^T tile (128 d x 32 keys)
      const int idx = threadIdx.x + i * 256;
      const int d = idx >> 2, kq = (idx & 3) * 8;
      async_g2l_b128(&Vsm[buf][d * 40 + kq],
                     &vt[vbase + (size_t)d * S + kb + kq]);
    }
  };

  v8f acc[8] = {};
  float m_r[8], l_r[8];
#pragma unroll
  for (int r = 0; r < 8; ++r) { m_r[r] = -1e30f; l_r[r] = 0.0f; }

  const int kend = q0b + 128;  // causal horizon for this block
  stage(0, 0);
  wait_async0();
  __syncthreads();

  int cur = 0;
  for (int kb = 0; kb < kend; kb += 32) {
    if (kb + 32 < kend) stage(kb + 32, cur ^ 1);  // DMA next tile now

    if (kb <= q0w + 15) {  // tile not fully masked for this wave
      v8f s0 = {}, s1 = {};
#pragma unroll
      for (int kc = 0; kc < 4; ++kc) {
        s0 = wmma_bf16(qf[kc], load_frag(&Ksm[cur][0 * 136 + kc * 32], 136), s0);
        s1 = wmma_bf16(qf[kc], load_frag(&Ksm[cur][16 * 136 + kc * 32], 136), s1);
      }
      const int qm_base = q0w + ((lane >> 4) << 3);
#pragma unroll
      for (int r = 0; r < 8; ++r) {
        const int qm = qm_base + r;
        const int kn0 = kb + (lane & 15);
        float a0 = s0[r] * scale, a1 = s1[r] * scale;
        if (kn0 > qm) a0 = -1e30f;
        if (kn0 + 16 > qm) a1 = -1e30f;
        s0[r] = a0; s1[r] = a1;
      }
      float alpha[8];
#pragma unroll
      for (int r = 0; r < 8; ++r) {
        const float mx = red_max16(fmaxf(s0[r], s1[r]));
        const float mn = fmaxf(m_r[r], mx);
        alpha[r] = __expf(m_r[r] - mn);
        m_r[r] = mn;
      }
#pragma unroll
      for (int r = 0; r < 8; ++r) {
        s0[r] = __expf(s0[r] - m_r[r]);
        s1[r] = __expf(s1[r] - m_r[r]);
        l_r[r] = l_r[r] * alpha[r] + red_sum16(s0[r] + s1[r]);
      }
#pragma unroll
      for (int dt = 0; dt < 8; ++dt)
#pragma unroll
        for (int r = 0; r < 8; ++r) acc[dt][r] *= alpha[r];

      // P: C-layout -> bf16 A-layout via per-wave LDS scratch.
#pragma unroll
      for (int r = 0; r < 8; ++r) {
        const int mrow = r + ((lane >> 4) << 3);
        pw[mrow * 40 + (lane & 15)] = (__bf16)s0[r];
        pw[mrow * 40 + 16 + (lane & 15)] = (__bf16)s1[r];
      }
      asm volatile("s_wait_dscnt 0" ::: "memory");
      const v16bf pf = load_frag(pw, 40);  // 16x32, K = 32 keys
#pragma unroll
      for (int dt = 0; dt < 8; ++dt)
        acc[dt] = wmma_bf16(pf, load_frag(&Vsm[cur][dt * 16 * 40], 40), acc[dt]);
    }
    wait_async0();
    __syncthreads();
    cur ^= 1;
  }

  // Normalize; store bf16 ctx in (B,S,HID) for the O-projection A-operand.
#pragma unroll
  for (int r = 0; r < 8; ++r) {
    const float inv_l = 1.0f / l_r[r];
    const int qrow = q0w + r + ((lane >> 4) << 3);
#pragma unroll
    for (int dt = 0; dt < 8; ++dt) {
      const int col = h * HD + dt * 16 + (lane & 15);
      ctx[((size_t)b * S + qrow) * HID + col] = (__bf16)(acc[dt][r] * inv_l);
    }
  }
}

// ---------------------------------------------------------------------------
extern "C" void kernel_launch(void* const* d_in, const int* in_sizes, int n_in,
                              void* d_out, int out_size, void* d_ws, size_t ws_size,
                              hipStream_t stream) {
  (void)in_sizes; (void)n_in; (void)out_size; (void)ws_size;
  const float* x     = (const float*)d_in[0];
  const float* Wq    = (const float*)d_in[1];
  const float* bq    = (const float*)d_in[2];
  const float* Wdown = (const float*)d_in[3];
  const float* Wk_up = (const float*)d_in[4];
  const float* Wv_up = (const float*)d_in[5];
  const float* Wo    = (const float*)d_in[6];
  const float* bo    = (const float*)d_in[7];
  float* out = (float*)d_out;

  char* ws = (char*)d_ws;
  const size_t MB = 1024ull * 1024ull;
  __bf16* x_bf   = (__bf16*)(ws + 0 * MB);    // 16 MB (B,S,HID)
  __bf16* WqT    = (__bf16*)(ws + 16 * MB);   //  8 MB (HID,HID)
  __bf16* WdownT = (__bf16*)(ws + 24 * MB);   //  2 MB (LAT,HID)
  __bf16* WkT    = (__bf16*)(ws + 26 * MB);   //  8 MB (HID,LAT)
  __bf16* WvT    = (__bf16*)(ws + 34 * MB);   //  8 MB (HID,LAT)
  __bf16* WoT    = (__bf16*)(ws + 42 * MB);   //  8 MB (HID,HID)
  __bf16* q_bf   = (__bf16*)(ws + 50 * MB);   // 16 MB (B,S,HID)
  __bf16* k_bf   = (__bf16*)(ws + 66 * MB);   // 16 MB (B,S,HID)
  __bf16* vT_bf  = (__bf16*)(ws + 82 * MB);   // 16 MB (B,NH,HD,S)
  __bf16* lat_bf = (__bf16*)(ws + 98 * MB);   //  4 MB (B,S,LAT)
  __bf16* ctx_bf = (__bf16*)(ws + 102 * MB);  // 16 MB (B,S,HID)

  const int M = B * S;  // 4096
  dim3 blk(256);

  // One-shot conversions / weight transposes.
  cvt_bf16_kernel<<<(B * S * HID / 4 + 255) / 256, blk, 0, stream>>>(
      x, x_bf, B * S * HID / 4);
  cvt_t_bf16_kernel<<<dim3(HID / 64, HID / 64), blk, 0, stream>>>(Wq, WqT, HID, HID);
  cvt_t_bf16_kernel<<<dim3(LAT / 64, HID / 64), blk, 0, stream>>>(Wdown, WdownT, HID, LAT);
  cvt_t_bf16_kernel<<<dim3(HID / 64, LAT / 64), blk, 0, stream>>>(Wk_up, WkT, LAT, HID);
  cvt_t_bf16_kernel<<<dim3(HID / 64, LAT / 64), blk, 0, stream>>>(Wv_up, WvT, LAT, HID);
  cvt_t_bf16_kernel<<<dim3(HID / 64, HID / 64), blk, 0, stream>>>(Wo, WoT, HID, HID);

  // Projections (bf16 out; V written pre-transposed).
  gemm_bf16_kernel<<<dim3(HID / 64, M / 128), blk, 0, stream>>>(
      x_bf, WqT, bq, q_bf, nullptr, M, HID, HID, 0);
  gemm_bf16_kernel<<<dim3(LAT / 64, M / 128), blk, 0, stream>>>(
      x_bf, WdownT, nullptr, lat_bf, nullptr, M, LAT, HID, 0);
  gemm_bf16_kernel<<<dim3(HID / 64, M / 128), blk, 0, stream>>>(
      lat_bf, WkT, nullptr, k_bf, nullptr, M, HID, LAT, 0);
  gemm_bf16_kernel<<<dim3(HID / 64, M / 128), blk, 0, stream>>>(
      lat_bf, WvT, nullptr, vT_bf, nullptr, M, HID, LAT, 1);

  // RoPE on q, k.
  rope_kernel<<<(B * S * NH * 16) / 256, blk, 0, stream>>>(q_bf, k_bf);

  // Causal flash attention.
  attn_kernel<<<dim3(B * NH, S / 128), blk, 0, stream>>>(q_bf, k_bf, vT_bf, ctx_bf);

  // Output projection (f32 out + bias).
  gemm_bf16_kernel<<<dim3(HID / 64, M / 128), blk, 0, stream>>>(
      ctx_bf, WoT, bo, nullptr, out, M, HID, HID, 2);
}
